// ResidualQuantizer_42803644072105
// MI455X (gfx1250) — matrix operand order repaired and checked by reading, not compiled, not run on hardware
//
#include <hip/hip_runtime.h>
#include <hip/hip_bf16.h>

#define NUM_LAYERS 4
#define K_CODES    2048
#define DIM        256
#define NROWS      32768
#define BETA       0.25f
#define EPSN       1e-12f

typedef __attribute__((ext_vector_type(16))) __bf16 v16bf;
typedef __attribute__((ext_vector_type(8)))  __bf16 v8bf;
typedef __attribute__((ext_vector_type(8)))  float  v8f;

// ---------------------------------------------------------------------------
// Normalize codebook rows -> bf16 normalized codebook (all 4 layers at once).
// ---------------------------------------------------------------------------
__global__ __launch_bounds__(256) void rq_normalize(const float* __restrict__ cb,
                                                    __bf16* __restrict__ cbn)
{
    const int lane = threadIdx.x & 31;
    const int row  = blockIdx.x * 8 + (threadIdx.x >> 5);     // 0 .. L*K-1
    const float* src = cb + (size_t)row * DIM + lane * 8;
    float4 a = *(const float4*)(src);
    float4 b = *(const float4*)(src + 4);
    float ss = a.x*a.x + a.y*a.y + a.z*a.z + a.w*a.w
             + b.x*b.x + b.y*b.y + b.z*b.z + b.w*b.w;
#pragma unroll
    for (int m = 16; m >= 1; m >>= 1) ss += __shfl_xor(ss, m, 32);
    const float scale = 1.0f / fmaxf(sqrtf(ss), EPSN);
    v8bf o;
    o[0] = (__bf16)(a.x*scale); o[1] = (__bf16)(a.y*scale);
    o[2] = (__bf16)(a.z*scale); o[3] = (__bf16)(a.w*scale);
    o[4] = (__bf16)(b.x*scale); o[5] = (__bf16)(b.y*scale);
    o[6] = (__bf16)(b.z*scale); o[7] = (__bf16)(b.w*scale);
    *(v8bf*)(cbn + (size_t)row * DIM + lane * 8) = o;
}

// ---------------------------------------------------------------------------
// residual(f32) = x ; residual(bf16) = x ; decoded = 0 ; loss = 0
// ---------------------------------------------------------------------------
__global__ __launch_bounds__(256) void rq_prep(const float* __restrict__ x,
                                               float* __restrict__ rf,
                                               __bf16* __restrict__ rb,
                                               float* __restrict__ dec,
                                               float* __restrict__ loss)
{
    const size_t i = (size_t)blockIdx.x * 256 + threadIdx.x;
    const float v = x[i];
    rf[i]  = v;
    rb[i]  = (__bf16)v;
    dec[i] = 0.0f;
    if (i == 0) *loss = 0.0f;
}

// ---------------------------------------------------------------------------
// Scores S = R(bf16) @ CBN^T via v_wmma_f32_16x16x32_bf16, fused row-argmax.
//
// Workgroup = 8 waves; each wave owns a 32-row tile (two resident A panels)
// so every 1KB B fragment read from LDS feeds TWO WMMAs (halves LDS
// bytes/FLOP, the binding resource). B fragments ping-pong through TWO
// buffers (16 VGPRs): batch s+1 loads issue before the WMMA pair of batch s,
// so each wait is partial (one batch in flight) while VGPRs stay < 256
// (no s_set_vgpr_msb, room for 2 workgroups per WGP).
// Codebook streamed through LDS with global_load_async_to_lds_b128
// (ASYNCcnt), double-buffered 64-code stages (32KB) shared by all 8 waves.
// LDS row stride 528B => half-wave b128 reads hit all 64 banks once.
// ---------------------------------------------------------------------------
#define STAGE_CODES 64
#define ROW_BYTES   512                    // 256 bf16
#define LDS_ROW     528                    // +16B pad (bank-conflict-free)
#define STAGE_BYTES (STAGE_CODES * LDS_ROW)

__device__ __forceinline__ v16bf rq_ldfrag(const char* brow, int s)
{
    const char* p = brow + 64 * s;
    v8bf lo = *(const v8bf*)(p);
    v8bf hh = *(const v8bf*)(p + 32);
    return __builtin_shufflevector(lo, hh, 0,1,2,3,4,5,6,7,
                                           8,9,10,11,12,13,14,15);
}

__global__ __launch_bounds__(256) void rq_gemm_argmax(const __bf16* __restrict__ Rbf,
                                                      const __bf16* __restrict__ cbn,
                                                      int* __restrict__ idx_out,
                                                      float* __restrict__ ids_f,
                                                      int layer)
{
    __shared__ __align__(16) unsigned char smem[2 * STAGE_BYTES];

    const int tid  = threadIdx.x;
    const int lane = tid & 31;
    const int wave = tid >> 5;
    const int row0 = (blockIdx.x * 8 + wave) * 32;   // 32 rows per wave
    const int nloc = lane & 15;
    const int hi   = lane >> 4;            // which half-wave
    const int kb   = hi * 8;               // A-fragment K offset per half

    // ---- two resident A panels: rows row0..+15 and row0+16..+31 ----
    const __bf16* Rrow0 = Rbf + (size_t)(row0 + nloc) * DIM;
    const __bf16* Rrow1 = Rrow0 + (size_t)16 * DIM;
    v16bf a0[8], a1[8];
#pragma unroll
    for (int s = 0; s < 8; ++s) {
        v8bf lo = *(const v8bf*)(Rrow0 + 32*s + kb);
        v8bf hh = *(const v8bf*)(Rrow0 + 32*s + 16 + kb);
        a0[s] = __builtin_shufflevector(lo, hh, 0,1,2,3,4,5,6,7,
                                                8,9,10,11,12,13,14,15);
        v8bf lo1 = *(const v8bf*)(Rrow1 + 32*s + kb);
        v8bf hh1 = *(const v8bf*)(Rrow1 + 32*s + 16 + kb);
        a1[s] = __builtin_shufflevector(lo1, hh1, 0,1,2,3,4,5,6,7,
                                                  8,9,10,11,12,13,14,15);
    }

    float bestv0[8], bestv1[8];
    int   besti0[8], besti1[8];
#pragma unroll
    for (int r = 0; r < 8; ++r) {
        bestv0[r] = -3.4e38f; besti0[r] = 0;
        bestv1[r] = -3.4e38f; besti1[r] = 0;
    }

    const char*    gsrc     = (const char*)cbn;                 // layer base
    const unsigned lds_base = (unsigned)(size_t)&smem[0];       // LDS byte offset

    // copy slots: chunk i handled by this thread = base + i*8 rows
    const unsigned lrow0 = (unsigned)((tid >> 5) * LDS_ROW   + (tid & 31) * 16);
    const unsigned grow0 = (unsigned)((tid >> 5) * ROW_BYTES + (tid & 31) * 16);

    // ---- prefetch stage 0 ----
#pragma unroll
    for (int i = 0; i < 8; ++i) {
        unsigned l = lds_base + lrow0 + i * (8 * LDS_ROW);
        unsigned g = grow0 + i * (8 * ROW_BYTES);
        asm volatile("global_load_async_to_lds_b128 %0, %1, %2"
                     :: "v"(l), "v"(g), "s"(gsrc) : "memory");
    }

#pragma unroll 1
    for (int st = 0; st < K_CODES / STAGE_CODES; ++st) {        // 32 stages
        asm volatile("s_wait_asynccnt 0" ::: "memory");
        __syncthreads();                                         // stage st ready

        if (st + 1 < K_CODES / STAGE_CODES) {                    // prefetch st+1
            const unsigned lbuf = lds_base + ((st + 1) & 1) * STAGE_BYTES + lrow0;
            const unsigned gofs = (unsigned)((st + 1) * STAGE_CODES * ROW_BYTES) + grow0;
#pragma unroll
            for (int i = 0; i < 8; ++i) {
                unsigned l = lbuf + i * (8 * LDS_ROW);
                unsigned g = gofs + i * (8 * ROW_BYTES);
                asm volatile("global_load_async_to_lds_b128 %0, %1, %2"
                             :: "v"(l), "v"(g), "s"(gsrc) : "memory");
            }
        }

        const char* bbuf = (const char*)&smem[(st & 1) * STAGE_BYTES];
#pragma unroll
        for (int t = 0; t < 4; ++t) {                            // 4 code tiles
            const char* brow = bbuf + (t * 16 + nloc) * LDS_ROW + hi * 16;

            v8f c0 = {}, c1 = {};
            v16bf bb0 = rq_ldfrag(brow, 0);                      // batch 0
            v16bf bb1;
#pragma unroll
            for (int s = 0; s < 8; ++s) {
                // issue next batch's loads BEFORE this batch's WMMAs
                if (s < 7) {
                    if ((s & 1) == 0) bb1 = rq_ldfrag(brow, s + 1);
                    else              bb0 = rq_ldfrag(brow, s + 1);
                }
                const v16bf cur = ((s & 1) == 0) ? bb0 : bb1;
                c0 = __builtin_amdgcn_wmma_f32_16x16x32_bf16(
                        false, a0[s], false, cur, (short)0, c0, false, false);
                c1 = __builtin_amdgcn_wmma_f32_16x16x32_bf16(
                        false, a1[s], false, cur, (short)0, c1, false, false);
                // bound live B fragments to the two ping-pong buffers
                asm volatile("" ::: "memory");
            }

            const int code = st * STAGE_CODES + t * 16 + nloc;
#pragma unroll
            for (int r = 0; r < 8; ++r) {                        // lane-local argmax
                float v0 = c0[r];
                if (v0 > bestv0[r]) { bestv0[r] = v0; besti0[r] = code; }
                float v1 = c1[r];
                if (v1 > bestv1[r]) { bestv1[r] = v1; besti1[r] = code; }
            }
        }
        __syncthreads();                                         // done reading st
    }

    // ---- cross-lane argmax within each 16-lane half; tie -> lowest index ----
#pragma unroll
    for (int r = 0; r < 8; ++r) {
        float v  = bestv0[r];
        int   ii = besti0[r];
#pragma unroll
        for (int m = 1; m < 16; m <<= 1) {
            float ov = __shfl_xor(v, m, 32);
            int   oi = __shfl_xor(ii, m, 32);
            if (ov > v || (ov == v && oi < ii)) { v = ov; ii = oi; }
        }
        float w  = bestv1[r];
        int   jj = besti1[r];
#pragma unroll
        for (int m = 1; m < 16; m <<= 1) {
            float ow = __shfl_xor(w, m, 32);
            int   oj = __shfl_xor(jj, m, 32);
            if (ow > w || (ow == w && oj < jj)) { w = ow; jj = oj; }
        }
        if (nloc == 0) {
            const int rowA = row0 + r + hi * 8;
            idx_out[rowA] = ii;
            ids_f[(size_t)rowA * NUM_LAYERS + layer] = (float)ii;
            const int rowB = row0 + 16 + r + hi * 8;
            idx_out[rowB] = jj;
            ids_f[(size_t)rowB * NUM_LAYERS + layer] = (float)jj;
        }
    }
}

// ---------------------------------------------------------------------------
// q = raw_cb[idx]; decoded += q; residual -= q (f32 + bf16 mirror);
// loss += ||residual_new||^2 (block reduce + atomic).
// ---------------------------------------------------------------------------
__global__ __launch_bounds__(256) void rq_update(const float* __restrict__ cb_layer,
                                                 const int* __restrict__ idx,
                                                 float* __restrict__ rf,
                                                 __bf16* __restrict__ rb,
                                                 float* __restrict__ dec,
                                                 float* __restrict__ loss)
{
    const size_t i  = (size_t)blockIdx.x * 256 + threadIdx.x;
    const int row   = (int)(i >> 8);
    const int d     = (int)(i & 255);
    const float q   = cb_layer[(size_t)idx[row] * DIM + d];
    const float rn  = rf[i] - q;
    rf[i]  = rn;
    rb[i]  = (__bf16)rn;
    dec[i] += q;

    float s = rn * rn;
#pragma unroll
    for (int m = 16; m >= 1; m >>= 1) s += __shfl_xor(s, m, 32);
    __shared__ float red[8];
    if ((threadIdx.x & 31) == 0) red[threadIdx.x >> 5] = s;
    __syncthreads();
    if (threadIdx.x == 0) {
        float t = red[0]+red[1]+red[2]+red[3]+red[4]+red[5]+red[6]+red[7];
        atomicAdd(loss, t);
    }
}

__global__ void rq_finalize(const float* __restrict__ loss, float* __restrict__ out_loss)
{
    *out_loss = (1.0f + BETA) * (*loss) / (float)((size_t)NROWS * DIM);
}

// ---------------------------------------------------------------------------
extern "C" void kernel_launch(void* const* d_in, const int* in_sizes, int n_in,
                              void* d_out, int out_size, void* d_ws, size_t ws_size,
                              hipStream_t stream)
{
    const float* x  = (const float*)d_in[0];   // [N, D]
    const float* cb = (const float*)d_in[1];   // [L, K, D]

    float* out      = (float*)d_out;
    float* out_ids  = out;                                         // N*L
    float* out_dec  = out + (size_t)NROWS * NUM_LAYERS;            // N*D
    float* out_loss = out_dec + (size_t)NROWS * DIM;               // 1

    char* ws = (char*)d_ws;
    float*  rf  = (float*)ws;   ws += (size_t)NROWS * DIM * sizeof(float);
    __bf16* rb  = (__bf16*)ws;  ws += (size_t)NROWS * DIM * sizeof(__bf16);
    __bf16* cbn = (__bf16*)ws;  ws += (size_t)NUM_LAYERS * K_CODES * DIM * sizeof(__bf16);
    int*    idxb = (int*)ws;    ws += (size_t)NROWS * sizeof(int);
    float*  lacc = (float*)ws;

    rq_normalize<<<NUM_LAYERS * K_CODES / 8, 256, 0, stream>>>(cb, cbn);
    rq_prep<<<(NROWS * DIM) / 256, 256, 0, stream>>>(x, rf, rb, out_dec, lacc);

    for (int l = 0; l < NUM_LAYERS; ++l) {
        rq_gemm_argmax<<<NROWS / 256, 256, 0, stream>>>(
            rb, cbn + (size_t)l * K_CODES * DIM, idxb, out_ids, l);
        rq_update<<<(NROWS * DIM) / 256, 256, 0, stream>>>(
            cb + (size_t)l * K_CODES * DIM, idxb, rf, rb, out_dec, lacc);
    }
    rq_finalize<<<1, 1, 0, stream>>>(lacc, out_loss);
}
